// RNN_27006754358125
// MI455X (gfx1250) — matrix operand order repaired and checked by reading, not compiled, not run on hardware
//
#include <hip/hip_runtime.h>
#include <hip/hip_bf16.h>

// ---------------------------------------------------------------------------
// LSTM (B=64, T=512, I=H=1024) for gfx1250 using v_wmma_f32_16x16x32_bf16.
//
// Phase 1: xg[t,b,g] = x[b,t,:] @ W_ih[g,:] + b_ih[g] + b_hh[g]   (big GEMM)
//          2 M-tiles x 4 N-tiles per wave -> 1.5 b128 loads per WMMA.
// Phase 2: 512 sequential step kernels:
//          gates = xg[t] + h @ W_hh^T ; LSTM cell update fused in-kernel.
// Weights pre-packed into WMMA B-fragment order (bf16) so the K-loop is
// two coalesced global_load_b128 per fragment per lane.
// ---------------------------------------------------------------------------

typedef __bf16 bf16_t;
typedef __attribute__((ext_vector_type(16))) __bf16 v16bf;
typedef __attribute__((ext_vector_type(8)))  __bf16 v8bf;
typedef __attribute__((ext_vector_type(8)))  float  v8f;

#define B_   64
#define T_   512
#define I_   1024
#define H_   1024
#define G_   4096          // 4*H
#define KT_  32            // K-tiles of 32 (K = 1024)
#define NT_  256           // N-tiles of 16 (N = 4096)

// ---- fragment loaders -----------------------------------------------------

__device__ __forceinline__ v8bf load8(const bf16_t* p) {
  return *(const v8bf*)p;   // 16B -> global_load_b128
}

// A fragment (16x32 bf16, row-major source, lda = 1024).
// ISA layout: lanes 0-15 row M=lane hold K=0..7 (v0..3) and K=16..23 (v4..7);
// lanes 16-31 row M=lane-16 hold K=8..15 and K=24..31.
__device__ __forceinline__ v16bf load_A(const bf16_t* rowptr, int k0, int half) {
  v8bf a0 = load8(rowptr + k0 + half * 8);
  v8bf a1 = load8(rowptr + k0 + half * 8 + 16);
  v16bf A;
#pragma unroll
  for (int i = 0; i < 8; ++i) { A[i] = a0[i]; A[8 + i] = a1[i]; }
  return A;
}

// B fragment (32x16 bf16) from fragment-major packed weights.
// Packed layout: [nt][kt][half(2)][lane(32)][8 bf16]  (fully coalesced).
__device__ __forceinline__ v16bf load_B(const bf16_t* __restrict__ pack,
                                        int nt, int kt, int lane) {
  const bf16_t* base = pack + ((size_t)(nt * KT_ + kt)) * 512;
  v8bf b0 = load8(base + lane * 8);
  v8bf b1 = load8(base + 256 + lane * 8);
  v16bf Bf;
#pragma unroll
  for (int i = 0; i < 8; ++i) { Bf[i] = b0[i]; Bf[8 + i] = b1[i]; }
  return Bf;
}

__device__ __forceinline__ v8f wmma_bf16(v16bf A, v16bf Bf, v8f C) {
  return __builtin_amdgcn_wmma_f32_16x16x32_bf16(
      false, A, false, Bf, (short)0, C, false, false);
}

// ---- prep kernels ---------------------------------------------------------

__global__ void cvt_bf16_kernel(const float* __restrict__ in,
                                bf16_t* __restrict__ out, int n) {
  int i = blockIdx.x * blockDim.x + threadIdx.x;
  if (i < n) out[i] = (bf16_t)in[i];
}

// Pack W[4096,1024] (row-major f32) into bf16 WMMA-B fragment order.
// One thread per packed bf16 element (4,194,304 threads).
__global__ void pack_w_kernel(const float* __restrict__ w,
                              bf16_t* __restrict__ out) {
  int tid  = blockIdx.x * blockDim.x + threadIdx.x;
  int e    = tid & 7;           // element within half-fragment
  int lane = (tid >> 3) & 31;
  int h    = (tid >> 8) & 1;    // half-fragment
  int kt   = (tid >> 9) & 31;
  int nt   = tid >> 14;
  int n = nt * 16 + (lane & 15);
  int k = kt * 32 + h * 16 + (lane >> 4) * 8 + e;   // B(k,n) = W[n,k]
  out[tid] = (bf16_t)w[(size_t)n * 1024 + k];
}

__global__ void init_state_kernel(bf16_t* __restrict__ h_a,
                                  bf16_t* __restrict__ h_b,
                                  float* __restrict__ c) {
  int i = blockIdx.x * blockDim.x + threadIdx.x;
  if (i < B_ * H_) { h_a[i] = (bf16_t)0.0f; h_b[i] = (bf16_t)0.0f; c[i] = 0.0f; }
}

// ---- phase 1: xg = x @ W_ih^T + (b_ih + b_hh) -----------------------------
// One wave computes a 32x64 block (2 M-tiles x 4 N-tiles) of xg,
// rows ordered (t*64+b). Per kt: 4 A-loads + 8 B-loads feed 8 WMMAs.
__global__ __launch_bounds__(128) void gemm_xg_kernel(
    const bf16_t* __restrict__ xb,    // [B,T,I] bf16
    const bf16_t* __restrict__ wp,    // packed W_ih fragments
    const float* __restrict__ b_ih, const float* __restrict__ b_hh,
    float* __restrict__ xg)           // [T,B,G] f32
{
  int wave = threadIdx.x >> 5;
  int lane = threadIdx.x & 31;
  int wid  = blockIdx.x * 4 + wave;      // 0 .. 65535
  int mtp  = wid >> 6;                   // M-tile pair 0..1023
  int ntg  = (wid & 63) << 2;            // first of 4 N-tiles
  int half = lane >> 4;

  const bf16_t* arow0;
  const bf16_t* arow1;
  {
    int r0 = (mtp * 2 + 0) * 16 + (lane & 15);
    int r1 = (mtp * 2 + 1) * 16 + (lane & 15);
    arow0 = xb + ((size_t)(r0 & 63) * T_ + (r0 >> 6)) * I_;
    arow1 = xb + ((size_t)(r1 & 63) * T_ + (r1 >> 6)) * I_;
  }

  v8f acc[2][4] = {};
  for (int kt = 0; kt < KT_; ++kt) {
    v16bf A0 = load_A(arow0, kt * 32, half);
    v16bf A1 = load_A(arow1, kt * 32, half);
#pragma unroll
    for (int j = 0; j < 4; ++j) {
      v16bf Bf = load_B(wp, ntg + j, kt, lane);
      acc[0][j] = wmma_bf16(A0, Bf, acc[0][j]);
      acc[1][j] = wmma_bf16(A1, Bf, acc[1][j]);
    }
  }

  // Epilogue: C layout -> N = lane&15, M = 8*half + v.
  int n = lane & 15;
#pragma unroll
  for (int i = 0; i < 2; ++i) {
    int mt  = mtp * 2 + i;
    int tt  = (mt * 16) >> 6;            // t constant within a tile
    int bb0 = (mt * 16) & 63;
#pragma unroll
    for (int j = 0; j < 4; ++j) {
      int g0 = (ntg + j) * 16 + n;
      float bias = b_ih[g0] + b_hh[g0];
#pragma unroll
      for (int v = 0; v < 8; ++v) {
        int bb = bb0 + 8 * half + v;
        xg[((size_t)tt * B_ + bb) * G_ + g0] = acc[i][j][v] + bias;
      }
    }
  }
}

// ---- phase 2: one timestep ------------------------------------------------
// grid = 256 blocks: blockIdx = batch-tile (0..3) + 4*hidden-tile (0..63).
// 4 waves/block, wave w computes gate w's 16x16 tile; fused cell update.
// Max wave parallelism here: this is the sequential latency-critical path.
__global__ __launch_bounds__(128) void lstm_step_kernel(
    const float* __restrict__ xg,     // [T,B,G]
    const bf16_t* __restrict__ wp,    // packed W_hh fragments
    const bf16_t* __restrict__ h_in,  // [B,H] bf16 (prev step)
    bf16_t* __restrict__ h_out,       // [B,H] bf16 (this step)
    float* __restrict__ c,            // [B,H] f32, in-place
    float* __restrict__ out,          // d_out [B,H] f32
    int t, int is_last)
{
  __shared__ float gsm[4 * 256];      // i/f/g/o 16x16 tiles

  int wave = threadIdx.x >> 5;
  int lane = threadIdx.x & 31;
  int half = lane >> 4;
  int mt = blockIdx.x & 3;            // batch tile
  int nh = blockIdx.x >> 2;           // hidden tile 0..63
  int nt = wave * 64 + nh;            // global N-tile for this gate

  const bf16_t* arow = h_in + (size_t)(mt * 16 + (lane & 15)) * H_;

  v8f acc = {};
  for (int kt = 0; kt < KT_; ++kt) {
    if (kt + 1 < KT_) {               // warm L2/WGP$ for next B fragment
      __builtin_prefetch(wp + ((size_t)(nt * KT_ + kt + 1)) * 512, 0, 3);
    }
    v16bf A  = load_A(arow, kt * 32, half);
    v16bf Bf = load_B(wp, nt, kt, lane);
    acc = wmma_bf16(A, Bf, acc);
  }

  // gates = acc + xg[t]; stage per-gate tiles in LDS
  int n = lane & 15;
  const float* xgt = xg + (size_t)t * B_ * G_;
#pragma unroll
  for (int v = 0; v < 8; ++v) {
    int m = 8 * half + v;
    float val = acc[v] + xgt[(size_t)(mt * 16 + m) * G_ + nt * 16 + n];
    gsm[wave * 256 + m * 16 + n] = val;
  }
  __syncthreads();

  // fused LSTM cell update: 256 elements, 128 threads -> 2 each
  for (int e = threadIdx.x; e < 256; e += 128) {
    float ig = 1.0f / (1.0f + __expf(-gsm[e]));
    float fg = 1.0f / (1.0f + __expf(-gsm[256 + e]));
    float gg = tanhf(gsm[512 + e]);
    float og = 1.0f / (1.0f + __expf(-gsm[768 + e]));
    int bglob = mt * 16 + (e >> 4);
    int hglob = nh * 16 + (e & 15);
    size_t ci = (size_t)bglob * H_ + hglob;
    float cn = fg * c[ci] + ig * gg;
    c[ci] = cn;
    float hn = og * tanhf(cn);
    h_out[ci] = (bf16_t)hn;
    if (is_last) out[ci] = hn;
  }
}

// ---- host side ------------------------------------------------------------

extern "C" void kernel_launch(void* const* d_in, const int* in_sizes, int n_in,
                              void* d_out, int out_size, void* d_ws, size_t ws_size,
                              hipStream_t stream) {
  (void)in_sizes; (void)n_in; (void)out_size; (void)ws_size;
  const float* x   = (const float*)d_in[0];
  const float* Wih = (const float*)d_in[1];
  const float* Whh = (const float*)d_in[2];
  const float* bih = (const float*)d_in[3];
  const float* bhh = (const float*)d_in[4];
  float* out = (float*)d_out;

  char* ws = (char*)d_ws;
  // workspace layout (all offsets 1MiB-aligned):
  bf16_t* xb   = (bf16_t*)(ws);                                   // 64 MiB
  bf16_t* wihp = (bf16_t*)(ws + ((size_t)64  << 20));             //  8 MiB
  bf16_t* whhp = (bf16_t*)(ws + ((size_t)72  << 20));             //  8 MiB
  bf16_t* h_a  = (bf16_t*)(ws + ((size_t)80  << 20));             // 128 KiB
  bf16_t* h_b  = (bf16_t*)(ws + ((size_t)81  << 20));             // 128 KiB
  float*  c    = (float*) (ws + ((size_t)82  << 20));             // 256 KiB
  float*  xg   = (float*) (ws + ((size_t)84  << 20));             // 512 MiB

  // 1) convert x to bf16
  {
    int n = B_ * T_ * I_;
    cvt_bf16_kernel<<<n / 256, 256, 0, stream>>>(x, xb, n);
  }
  // 2) pack weights into WMMA fragment order
  {
    int n = NT_ * KT_ * 512;          // 4,194,304 packed elements
    pack_w_kernel<<<n / 256, 256, 0, stream>>>(Wih, wihp);
    pack_w_kernel<<<n / 256, 256, 0, stream>>>(Whh, whhp);
  }
  // 3) init h0 = c0 = 0
  init_state_kernel<<<(B_ * H_) / 256, 256, 0, stream>>>(h_a, h_b, c);

  // 4) phase 1: xg = x @ W_ih^T + bias   (65536 waves / 16384 blocks)
  gemm_xg_kernel<<<16384, 128, 0, stream>>>(xb, wihp, bih, bhh, xg);

  // 5) phase 2: 512 sequential steps, h double-buffered
  for (int t = 0; t < T_; ++t) {
    const bf16_t* hin  = (t & 1) ? h_b : h_a;
    bf16_t*       hout = (t & 1) ? h_a : h_b;
    lstm_step_kernel<<<256, 128, 0, stream>>>(
        xg, whhp, hin, hout, c, out, t, (t == T_ - 1) ? 1 : 0);
  }
}
